// UserGraphAttention_57801669869933
// MI455X (gfx1250) — matrix operand order repaired and checked by reading, not compiled, not run on hardware
//
#include <hip/hip_runtime.h>
#include <hip/hip_bf16.h>
#include <math.h>

#define D_MODEL 512
#define NBATCH  8192
#define LN_EPS  1e-5f

typedef __bf16 bf16;
typedef __attribute__((ext_vector_type(16))) __bf16 v16bf;
typedef __attribute__((ext_vector_type(8)))  __bf16 v8bf;
typedef __attribute__((ext_vector_type(8)))  float  v8f;

// ---------------------------------------------------------------------------
// WMMA helper: D = A(16x32 bf16) * B(32x16 bf16) + C(16x16 f32)
// ---------------------------------------------------------------------------
__device__ __forceinline__ v8f wmma_bf16(v16bf a, v16bf b, v8f c) {
  return __builtin_amdgcn_wmma_f32_16x16x32_bf16(
      /*neg_a=*/false, a, /*neg_b=*/false, b,
      /*c_mod=*/(short)0, c, /*reuse_a=*/false, /*reuse_b=*/false);
}

// A fragment: 16x32 bf16 tile, row-major source, row m = lane&15.
__device__ __forceinline__ v16bf load_frag_a(const bf16* base, int ld, int m0,
                                             int k0, int lane) {
  const int mr = lane & 15, half = lane >> 4;
  const bf16* p = base + (size_t)(m0 + mr) * ld + k0 + half * 16;
  v8bf lo = *(const v8bf*)(p);
  v8bf hi = *(const v8bf*)(p + 8);
  return __builtin_shufflevector(lo, hi, 0, 1, 2, 3, 4, 5, 6, 7,
                                 8, 9, 10, 11, 12, 13, 14, 15);
}

// B fragment: 32x16 bf16 (KxN) taken as transposed rows of a row-major source.
__device__ __forceinline__ v16bf load_frag_b(const bf16* base, int ld, int n0,
                                             int k0, int lane) {
  const int nc = lane & 15, half = lane >> 4;
  const bf16* p = base + (size_t)(n0 + nc) * ld + k0 + half * 16;
  return *(const v16bf*)(p);
}

// B fragment from an LDS-staged 16x64 chunk (row stride 64 bf16).
__device__ __forceinline__ v16bf load_frag_b_lds(const bf16* buf, int k0,
                                                 int lane) {
  const int nc = lane & 15, half = lane >> 4;
  const bf16* p = buf + nc * 64 + k0 + half * 16;
  return *(const v16bf*)(p);
}

// ---------------------------------------------------------------------------
// Async global->LDS staging of one 16x64 bf16 K-chunk (2 KB) into a per-wave
// LDS buffer.  4 x global_load_async_to_lds_b128 (32 lanes x 16B = 4 rows per
// op).  ASYNCcnt-tracked; s_wait_dscnt first to close the WAR window against
// this wave's own ds reads of the buffer being recycled.
// NOTE: LDS byte offsets assume the single static __shared__ block is at LDS
// offset 0 (standard placement for a single shared object).
// ---------------------------------------------------------------------------
__device__ __forceinline__ void stage_k_chunk(const bf16* __restrict__ Kb,
                                              int krow0, int kc,
                                              unsigned buf_off, int lane) {
  const int r  = lane >> 3;  // 0..3  (row within group of 4)
  const int cq = lane & 7;   // 0..7  (16B column chunk)
  asm volatile("s_wait_dscnt 0x0" ::: "memory");
#pragma unroll
  for (int j = 0; j < 4; ++j) {
    const bf16* g = Kb + (size_t)(krow0 + j * 4 + r) * D_MODEL + kc + cq * 8;
    unsigned l = buf_off + (unsigned)((j * 4 + r) * 128 + cq * 16);
    asm volatile("global_load_async_to_lds_b128 %0, %1, off"
                 :: "v"(l), "v"(g) : "memory");
  }
}

// Score tile for K-tile tn: double-buffered async staging (64-wide K chunks)
// feeding v_wmma_f32_16x16x32_bf16 from LDS.
__device__ __forceinline__ v8f score_tile_async(
    const bf16* __restrict__ Kb, const v16bf* qfr,
    const unsigned char* smembase, unsigned mybase, int tn, int lane) {
  stage_k_chunk(Kb, tn * 16, 0, mybase, lane);
  v8f c = {};
#pragma unroll
  for (int ch = 0; ch < 8; ++ch) {
    if (ch < 7) {
      stage_k_chunk(Kb, tn * 16, (ch + 1) * 64,
                    mybase + (unsigned)(((ch + 1) & 1) * 2048), lane);
      asm volatile("s_wait_asynccnt 0x4" ::: "memory");  // chunk ch landed
    } else {
      asm volatile("s_wait_asynccnt 0x0" ::: "memory");
    }
    const bf16* bb = (const bf16*)(smembase + mybase + (unsigned)((ch & 1) * 2048));
    c = wmma_bf16(qfr[2 * ch + 0], load_frag_b_lds(bb, 0, lane), c);
    c = wmma_bf16(qfr[2 * ch + 1], load_frag_b_lds(bb, 32, lane), c);
  }
  return c;
}

// ---------------------------------------------------------------------------
// f32 -> bf16 conversion
// ---------------------------------------------------------------------------
__global__ __launch_bounds__(256) void convert_f32_bf16(
    const float* __restrict__ in, bf16* __restrict__ out, int n) {
  int i = blockIdx.x * blockDim.x + threadIdx.x;
  if (i < n) out[i] = (bf16)in[i];
}

// ---------------------------------------------------------------------------
// O[M,N] = A[M,512](bf16) @ W^T, one 16x16 tile per wave, K chunks of 32.
// ---------------------------------------------------------------------------
template <bool OUT_BF16>
__global__ __launch_bounds__(256) void gemm_xw_kernel(
    const bf16* __restrict__ A, const bf16* __restrict__ W,
    void* __restrict__ Out) {
  const int lane = threadIdx.x & 31;
  const int wave = threadIdx.x >> 5;
  const int w = blockIdx.x * 8 + wave;
  const int tn = w & (D_MODEL / 16 - 1);
  const int tm = w >> 5;
  const int m0 = tm * 16, n0 = tn * 16;

  v8f c = {};
#pragma unroll
  for (int k0 = 0; k0 < D_MODEL; k0 += 32) {
    v16bf a = load_frag_a(A, D_MODEL, m0, k0, lane);
    v16bf b = load_frag_b(W, D_MODEL, n0, k0, lane);
    c = wmma_bf16(a, b, c);
  }

  const int col = lane & 15, half = lane >> 4;
  if (OUT_BF16) {
    bf16* o = (bf16*)Out;
#pragma unroll
    for (int i = 0; i < 8; ++i)
      o[(size_t)(m0 + i + 8 * half) * D_MODEL + n0 + col] = (bf16)c[i];
  } else {
    float* o = (float*)Out;
#pragma unroll
    for (int i = 0; i < 8; ++i)
      o[(size_t)(m0 + i + 8 * half) * D_MODEL + n0 + col] = c[i];
  }
}

// ---------------------------------------------------------------------------
// Fused sparse-attention kernel: one 256-thread block per 16-row strip.
// LDS regions aliased by lifetime (total 61952 B static):
//   [     0, 2560) header (top32 lists, thresholds, counters, LN stats)
//   [ 2560,10752) nidx/nval neighbor lists        (pass2 .. PV)
//   [10752,27136) oab bf16 attn@V                 (PV .. Wo)
//                 `-- aliased by cand[16][128]    (pass1 only)
//   [27136,59904) hbuf f32 residual               (Wo .. LN)
//                 `-- aliased by K staging 8x2x2KB (pass1+pass2 only)
//   [59904,61952) LN reduction scratch
// ---------------------------------------------------------------------------
__global__ __launch_bounds__(256) void attn_kernel(
    const bf16* __restrict__ Qb, const bf16* __restrict__ Kb,
    const float* __restrict__ V, const bf16* __restrict__ Wob,
    const float* __restrict__ x, const float* __restrict__ bo,
    const float* __restrict__ gamma, const float* __restrict__ beta,
    float* __restrict__ out) {
  __shared__ __align__(16) unsigned char smem[61952];
  float* top    = (float*)(smem + 0);      // [16][32] sorted desc
  float* thrArr = (float*)(smem + 2048);   // [16]
  int*   cnt    = (int*)  (smem + 2112);   // [16]
  float* rowmax = (float*)(smem + 2176);   // [16]
  float* rsum   = (float*)(smem + 2240);   // [16] (1/sum)
  int*   ncnt   = (int*)  (smem + 2304);   // [16]
  float* muArr  = (float*)(smem + 2368);   // [16]
  float* rsArr  = (float*)(smem + 2432);   // [16]
  int*   nidx   = (int*)  (smem + 2560);   // [16][64]
  float* nval   = (float*)(smem + 6656);   // [16][64]
  bf16*  oab    = (bf16*) (smem + 10752);  // [16][512]   (aliases cand)
  float* cand   = (float*)(smem + 10752);  // [16][128]   pass-1 only
  float* hbuf   = (float*)(smem + 27136);  // [16][512]   (aliases kstage)
  float* redS   = (float*)(smem + 59904);  // [256]
  float* redQ   = (float*)(smem + 60928);  // [256]
  const unsigned KSTAGE = 27136;           // 8 waves x 2 bufs x 2048 B

  const int tid  = threadIdx.x;
  const int lane = tid & 31;
  const int wave = tid >> 5;
  const int m0   = blockIdx.x * 16;
  const float scale = 0.044194173824159216f;  // 512^-0.5
  const unsigned mybase = KSTAGE + (unsigned)wave * 4096u;

  // Prefetch the residual x strip (consumed in the Wo phase).
  __builtin_prefetch(x + (size_t)(m0 + (tid & 15)) * D_MODEL + (tid >> 4) * 32,
                     0, 3);

  for (int i = tid; i < 16 * 32; i += 256) top[i] = -1e30f;
  if (tid < 16) { thrArr[tid] = -1e30f; cnt[tid] = 0; ncnt[tid] = 0; }
  __syncthreads();

  // Whole 16x512 bf16 Q strip resident in VGPRs (128 VGPRs/lane).
  v16bf qfr[16];
#pragma unroll
  for (int kk = 0; kk < 16; ++kk)
    qfr[kk] = load_frag_a(Qb, D_MODEL, m0, kk * 32, lane);

  const int chalf = lane >> 4;
  const int ccol  = lane & 15;

  // ---- Pass 1: top-32 threshold per row ----
  for (int t0 = 0; t0 < NBATCH / 16; t0 += 8) {
    const int tn = t0 + wave;
    v8f c = score_tile_async(Kb, qfr, smem, mybase, tn, lane);
#pragma unroll
    for (int i = 0; i < 8; ++i) {
      const int row = i + 8 * chalf;
      const float s = c[i] * scale;
      if (s > thrArr[row]) {
        int slot = atomicAdd(&cnt[row], 1);
        if (slot < 128) cand[row * 128 + slot] = s;
      }
    }
    __syncthreads();
    if (tid < 16) {  // serial sorted-insert merge, one thread per row
      const int r = tid;
      int nc2 = cnt[r]; if (nc2 > 128) nc2 = 128;
      for (int j = 0; j < nc2; ++j) {
        float v = cand[r * 128 + j];
        if (v > top[r * 32 + 31]) {
          int p = 31;
          while (p > 0 && top[r * 32 + p - 1] < v) {
            top[r * 32 + p] = top[r * 32 + p - 1]; --p;
          }
          top[r * 32 + p] = v;
        }
      }
      cnt[r] = 0;
      thrArr[r] = top[r * 32 + 31];
    }
    __syncthreads();
  }

  if (tid < 16) { rowmax[tid] = top[tid * 32]; thrArr[tid] = top[tid * 32 + 31]; }
  __syncthreads();

  // ---- Pass 2: gather neighbors >= threshold ----
  for (int t0 = 0; t0 < NBATCH / 16; t0 += 8) {
    const int tn = t0 + wave;
    v8f c = score_tile_async(Kb, qfr, smem, mybase, tn, lane);
#pragma unroll
    for (int i = 0; i < 8; ++i) {
      const int row = i + 8 * chalf;
      const float s = c[i] * scale;
      if (s >= thrArr[row]) {
        int slot = atomicAdd(&ncnt[row], 1);
        if (slot < 64) {
          nidx[row * 64 + slot] = tn * 16 + ccol;
          nval[row * 64 + slot] = __expf(s - rowmax[row]);
        }
      }
    }
  }
  __syncthreads();
  if (tid < 16) {
    const int r = tid;
    int m = ncnt[r]; if (m > 64) m = 64;
    ncnt[r] = m;
    float sum = 0.f;
    for (int j = 0; j < m; ++j) sum += nval[r * 64 + j];
    rsum[r] = 1.0f / sum;
  }
  __syncthreads();

  // ---- Sparse PV: thread (r = tid>>4, g = tid&15) owns 32 f32 accumulators
  {
    const int r = tid >> 4, g = tid & 15;
    float acc[32];
#pragma unroll
    for (int u = 0; u < 32; ++u) acc[u] = 0.f;
    const int m = ncnt[r];
    const float inv = rsum[r];
    for (int j = 0; j < m; ++j) {
      const float w = nval[r * 64 + j] * inv;
      const float* vrow = V + (size_t)nidx[r * 64 + j] * D_MODEL + g * 32;
#pragma unroll
      for (int u = 0; u < 32; ++u) acc[u] += w * vrow[u];
    }
#pragma unroll
    for (int u = 0; u < 32; ++u)
      oab[r * D_MODEL + g * 32 + u] = (bf16)acc[u];
  }
  __syncthreads();

  // ---- Wo projection via WMMA (A fragments from LDS), bias + residual ----
  v16bf ofr[16];
#pragma unroll
  for (int kk = 0; kk < 16; ++kk)
    ofr[kk] = load_frag_a((const bf16*)oab, D_MODEL, 0, kk * 32, lane);

  for (int tn = wave; tn < D_MODEL / 16; tn += 8) {
    v8f c = {};
#pragma unroll
    for (int kk = 0; kk < 16; ++kk) {
      v16bf b = load_frag_b(Wob, D_MODEL, tn * 16, kk * 32, lane);
      c = wmma_bf16(ofr[kk], b, c);
    }
#pragma unroll
    for (int i = 0; i < 8; ++i) {
      const int row = i + 8 * chalf;
      const int gc  = tn * 16 + ccol;
      hbuf[row * D_MODEL + gc] =
          c[i] + bo[gc] + x[(size_t)(m0 + row) * D_MODEL + gc];
    }
  }
  __syncthreads();

  // ---- LayerNorm ----
  {
    const int r = tid >> 4, g = tid & 15;
    float s = 0.f, q = 0.f;
#pragma unroll
    for (int u = 0; u < 32; ++u) {
      float v = hbuf[r * D_MODEL + g * 32 + u];
      s += v; q += v * v;
    }
    redS[tid] = s; redQ[tid] = q;
  }
  __syncthreads();
  if (tid < 16) {
    float s = 0.f, q = 0.f;
    for (int g = 0; g < 16; ++g) { s += redS[tid * 16 + g]; q += redQ[tid * 16 + g]; }
    const float mu  = s / (float)D_MODEL;
    const float var = q / (float)D_MODEL - mu * mu;
    muArr[tid] = mu;
    rsArr[tid] = rsqrtf(var + LN_EPS);
  }
  __syncthreads();
  {
    const int r = tid >> 4, g = tid & 15;
    const float mu = muArr[r], rs = rsArr[r];
#pragma unroll
    for (int u = 0; u < 32; ++u) {
      const int cix = g * 32 + u;
      out[(size_t)(m0 + r) * D_MODEL + cix] =
          (hbuf[r * D_MODEL + cix] - mu) * rs * gamma[cix] + beta[cix];
    }
  }
}

// ---------------------------------------------------------------------------
extern "C" void kernel_launch(void* const* d_in, const int* in_sizes, int n_in,
                              void* d_out, int out_size, void* d_ws,
                              size_t ws_size, hipStream_t stream) {
  const float* x  = (const float*)d_in[0];
  const float* Wq = (const float*)d_in[1];
  const float* Wk = (const float*)d_in[2];
  const float* Wv = (const float*)d_in[3];
  const float* Wo = (const float*)d_in[4];
  const float* bo = (const float*)d_in[5];
  const float* gm = (const float*)d_in[6];
  const float* bt = (const float*)d_in[7];
  float* out = (float*)d_out;

  char* w = (char*)d_ws;
  bf16* xb  = (bf16*)w; w += (size_t)NBATCH * D_MODEL * 2;
  bf16* Qb  = (bf16*)w; w += (size_t)NBATCH * D_MODEL * 2;
  bf16* Kb  = (bf16*)w; w += (size_t)NBATCH * D_MODEL * 2;
  bf16* Wqb = (bf16*)w; w += (size_t)D_MODEL * D_MODEL * 2;
  bf16* Wkb = (bf16*)w; w += (size_t)D_MODEL * D_MODEL * 2;
  bf16* Wvb = (bf16*)w; w += (size_t)D_MODEL * D_MODEL * 2;
  bf16* Wob = (bf16*)w; w += (size_t)D_MODEL * D_MODEL * 2;
  float* V  = (float*)w; w += (size_t)NBATCH * D_MODEL * 4;

  const int nx = NBATCH * D_MODEL;
  const int nw = D_MODEL * D_MODEL;
  convert_f32_bf16<<<(nx + 255) / 256, 256, 0, stream>>>(x,  xb,  nx);
  convert_f32_bf16<<<(nw + 255) / 256, 256, 0, stream>>>(Wq, Wqb, nw);
  convert_f32_bf16<<<(nw + 255) / 256, 256, 0, stream>>>(Wk, Wkb, nw);
  convert_f32_bf16<<<(nw + 255) / 256, 256, 0, stream>>>(Wv, Wvb, nw);
  convert_f32_bf16<<<(nw + 255) / 256, 256, 0, stream>>>(Wo, Wob, nw);

  gemm_xw_kernel<true ><<<2048, 256, 0, stream>>>(xb, Wqb, (void*)Qb);
  gemm_xw_kernel<true ><<<2048, 256, 0, stream>>>(xb, Wkb, (void*)Kb);
  gemm_xw_kernel<false><<<2048, 256, 0, stream>>>(xb, Wvb, (void*)V);

  attn_kernel<<<NBATCH / 16, 256, 0, stream>>>(Qb, Kb, V, Wob, x, bo, gm, bt,
                                               out);
}
